// Head_1486058684718
// MI455X (gfx1250) — matrix operand order repaired and checked by reading, not compiled, not run on hardware
//
#include <hip/hip_runtime.h>
#include <stdint.h>

typedef __attribute__((ext_vector_type(16))) __bf16 v16bf;
typedef __attribute__((ext_vector_type(8)))  __bf16 v8bf;
typedef __attribute__((ext_vector_type(8)))  float  v8f;
typedef __attribute__((ext_vector_type(4)))  float  f32x4;

#define B_ 4
#define T_ 4096
#define E_ 1024
#define H_ 64
#define BT_ (B_ * T_)

#define KROW 72   // padded K-tile row pitch in bf16 (144B: 16B aligned, bank-clean)
#define VROW 40   // padded V/W-tile row pitch in bf16 (80B: 16B aligned, bank-clean)

static __device__ __forceinline__ v8f wmma_bf16(v16bf a, v16bf b, v8f c) {
    return __builtin_amdgcn_wmma_f32_16x16x32_bf16(
        false, a, false, b, (short)0, c, false, false);
}

// gfx1250 async global->LDS copy (ASYNCcnt path), 16 bytes per lane.
static __device__ __forceinline__ void async_cp16(unsigned lds_byte_addr,
                                                  const void* gptr) {
    asm volatile("global_load_async_to_lds_b128 %0, %1, off"
                 :: "v"(lds_byte_addr),
                    "v"((unsigned long long)(uintptr_t)gptr)
                 : "memory");
}
#define WAIT_ASYNC_ALL() asm volatile("s_wait_asynccnt 0" ::: "memory")
#define WAIT_ASYNC_2()   asm volatile("s_wait_asynccnt 2" ::: "memory")
#define WAIT_ASYNC_3()   asm volatile("s_wait_asynccnt 3" ::: "memory")

static __device__ __forceinline__ unsigned lds_addr(const void* p) {
    // low 32 bits of a generic LDS pointer == LDS byte offset
    return (unsigned)(uintptr_t)p;
}

// ---------------------------------------------------------------------------
// Kernel 0: transpose + convert weights to bf16: Wt[h][e] = W[e][h]
// Fold attention scale 1/sqrt(H)=0.125 into Wq.
// ---------------------------------------------------------------------------
__global__ void prep_weights(const float* __restrict__ Wk,
                             const float* __restrict__ Wq,
                             const float* __restrict__ Wv,
                             __bf16* __restrict__ wtk,
                             __bf16* __restrict__ wtq,
                             __bf16* __restrict__ wtv) {
    int tid = blockIdx.x * 256 + threadIdx.x;       // 0 .. 3*E*H-1
    int mat = tid / (E_ * H_);
    int r   = tid % (E_ * H_);
    int e = r / H_, h = r % H_;
    const float* W  = (mat == 0) ? Wk : (mat == 1) ? Wq : Wv;
    __bf16*      Wt = (mat == 0) ? wtk : (mat == 1) ? wtq : wtv;
    float v = W[(size_t)e * H_ + h];
    if (mat == 1) v *= 0.125f;                      // fold softmax scale into Q
    Wt[(size_t)h * E_ + e] = (__bf16)v;
}

// ---------------------------------------------------------------------------
// Kernel 1: fused QKV projection.  8 waves/block, 16 rows/wave.  Weight
// K-chunks are async-staged into LDS once per block (8x traffic reduction),
// double-buffered.  q,k -> bf16 [BT,H]; v -> bf16 transposed [B,H,T].
// ---------------------------------------------------------------------------
__global__ void qkv_proj(const float*  __restrict__ x,
                         const __bf16* __restrict__ wtq,
                         const __bf16* __restrict__ wtk,
                         const __bf16* __restrict__ wtv,
                         __bf16* __restrict__ qo,
                         __bf16* __restrict__ ko,
                         __bf16* __restrict__ vto) {
    __shared__ __attribute__((aligned(16))) __bf16 wsm[2][3][64 * VROW];

    const int tid  = threadIdx.x;
    const int lane = tid & 31;
    const int wave = tid >> 5;
    const int l15  = lane & 15;
    const int h2   = lane >> 4;
    const int tbase = (blockIdx.x * 8 + wave) * 16;

    const __bf16* wts[3] = {wtq, wtk, wtv};

    // stage mapping: thread -> (h row, 16B chunk), one chunk per matrix
    const int sh = tid >> 2;          // 0..63
    const int sc = tid & 3;           // 0..3  (4 x 16B = 32 bf16 per row)
    auto stage = [&](int it) {
        const int buf = it & 1;
        const int eb  = it * 32;
#pragma unroll
        for (int m = 0; m < 3; ++m) {
            unsigned l = lds_addr(&wsm[buf][m][sh * VROW]) + sc * 16;
            const char* g = (const char*)(wts[m] + (size_t)sh * E_ + eb) + sc * 16;
            async_cp16(l, g);
        }
    };

    v8f acc[3][4];
#pragma unroll
    for (int m = 0; m < 3; ++m)
#pragma unroll
        for (int n = 0; n < 4; ++n) acc[m][n] = (v8f)0.0f;

    const float* xrow = x + (size_t)(tbase + l15) * E_;

    stage(0);
    const int NIT = E_ / 32;
    for (int it = 0; it < NIT; ++it) {
        const int eb = it * 32;
        // ---- A operand: 16x32 slice of x (wave-private, global path) ----
        int e0 = eb + 16 * h2;
        f32x4 xa0 = *(const f32x4*)(xrow + e0);
        f32x4 xa1 = *(const f32x4*)(xrow + e0 + 4);
        f32x4 xb0 = *(const f32x4*)(xrow + e0 + 8);
        f32x4 xb1 = *(const f32x4*)(xrow + e0 + 12);
        v16bf a;
#pragma unroll
        for (int j = 0; j < 4; ++j) {
            a[j]      = (__bf16)xa0[j];
            a[4 + j]  = (__bf16)xa1[j];
            a[8 + j]  = (__bf16)xb0[j];
            a[12 + j] = (__bf16)xb1[j];
        }

        if (it + 1 < NIT) { stage(it + 1); WAIT_ASYNC_3(); }
        else              { WAIT_ASYNC_ALL(); }
        __syncthreads();                      // staged weights visible

        const int buf = it & 1;
#pragma unroll
        for (int m = 0; m < 3; ++m) {
#pragma unroll
            for (int nt = 0; nt < 4; ++nt) {
                const __bf16* bp = &wsm[buf][m][(16 * nt + l15) * VROW + 16 * h2];
                v16bf bv = *(const v16bf*)bp;
                acc[m][nt] = wmma_bf16(a, bv, acc[m][nt]);
            }
        }
        __syncthreads();                      // reads done before overwrite
    }

    // ---- stores ----
    const int b  = tbase / T_;                // tiles never straddle batches
    const int ti = (tbase % T_) + 8 * h2;
#pragma unroll
    for (int nt = 0; nt < 4; ++nt) {
        int hcol = 16 * nt + l15;
#pragma unroll
        for (int r = 0; r < 8; ++r) {
            size_t t = (size_t)(tbase + r + 8 * h2);
            qo[t * H_ + hcol] = (__bf16)acc[0][nt][r];   // scaled q (via Wq)
            ko[t * H_ + hcol] = (__bf16)acc[1][nt][r];
        }
        v8bf pv;
#pragma unroll
        for (int r = 0; r < 8; ++r) pv[r] = (__bf16)acc[2][nt][r];
        *(v8bf*)(vto + ((size_t)b * H_ + hcol) * T_ + ti) = pv;   // v^T
    }
}

// ---------------------------------------------------------------------------
// Kernel 2: flash attention.  8 waves/block = 8 query tiles sharing K/V
// blocks async-staged into LDS (double-buffered).  S^T = K.Q^T so exp(S-m)
// is already in the A-operand layout for P.V (no cross-lane transpose).
// ---------------------------------------------------------------------------
__global__ void flash_attn(const __bf16* __restrict__ q,
                           const __bf16* __restrict__ k,
                           const __bf16* __restrict__ vt,
                           float* __restrict__ out) {
    __shared__ __attribute__((aligned(16))) __bf16 ksm[2][32 * KROW];
    __shared__ __attribute__((aligned(16))) __bf16 vsm[2][64 * VROW];

    const int tid  = threadIdx.x;
    const int lane = tid & 31;
    const int wave = tid >> 5;
    const int l15  = lane & 15;
    const int h2   = lane >> 4;

    const int b     = blockIdx.x >> 5;              // 32 blocks per batch
    const int tile  = (blockIdx.x & 31) * 8 + wave; // query tile in batch
    const int q0    = tile * 16;
    const size_t g0 = (size_t)b * T_ + q0;

    const char* kgb = (const char*)(k  + (size_t)b * T_ * H_);
    const char* vgb = (const char*)(vt + (size_t)b * H_ * T_);

    // stage mapping: K tile 32x64 bf16 (8 chunks/row), V^T tile 64x32 (4/row)
    const int kr = tid >> 3, kc = tid & 7;
    const int vh = tid >> 2, vc = tid & 3;
    auto stage = [&](int it) {
        const int buf = it & 1;
        const int kb  = it * 32;
        async_cp16(lds_addr(&ksm[buf][kr * KROW]) + kc * 16,
                   kgb + ((size_t)(kb + kr) * H_) * 2 + kc * 16);
        async_cp16(lds_addr(&vsm[buf][vh * VROW]) + vc * 16,
                   vgb + ((size_t)vh * T_ + kb) * 2 + vc * 16);
    };

    // Q as B operand (64x16): lane = query col, elems contiguous over h
    const __bf16* qrow = q + (g0 + l15) * H_;
    v16bf qb0 = *(const v16bf*)(qrow + 16 * h2);         // h in [0,32)
    v16bf qb1 = *(const v16bf*)(qrow + 32 + 16 * h2);    // h in [32,64)

    v8f O[4];
#pragma unroll
    for (int n = 0; n < 4; ++n) O[n] = (v8f)0.0f;
    float mrow = -3.0e38f;        // running row max (one query per lane)
    float lrow = 0.0f;            // running row sum

    stage(0);
    const int NIT = T_ / 32;
    for (int it = 0; it < NIT; ++it) {
        if (it + 1 < NIT) { stage(it + 1); WAIT_ASYNC_2(); }
        else              { WAIT_ASYNC_ALL(); }
        __syncthreads();                      // staged K/V visible
        const int buf = it & 1;

        // ---- K as A operand from LDS: 2 key sub-tiles x 2 h-chunks ----
        v16bf ka[2][2];
#pragma unroll
        for (int t2 = 0; t2 < 2; ++t2) {
            const __bf16* krow = &ksm[buf][(16 * t2 + l15) * KROW];
#pragma unroll
            for (int c = 0; c < 2; ++c) {
                const __bf16* p0 = krow + 32 * c + 16 * h2;
                v8bf lo = *(const v8bf*)p0;
                v8bf hi = *(const v8bf*)(p0 + 8);
                v16bf av;
#pragma unroll
                for (int j = 0; j < 8; ++j) { av[j] = lo[j]; av[8 + j] = hi[j]; }
                ka[t2][c] = av;
            }
        }
        // S^T tiles: s0 = keys[kb..kb+15], s1 = keys[kb+16..kb+31]
        v8f s0 = (v8f)0.0f, s1 = (v8f)0.0f;
        s0 = wmma_bf16(ka[0][0], qb0, s0);
        s0 = wmma_bf16(ka[0][1], qb1, s0);
        s1 = wmma_bf16(ka[1][0], qb0, s1);
        s1 = wmma_bf16(ka[1][1], qb1, s1);

        // ---- online softmax: each lane owns query l15; partner = lane^16 ----
        float mx = s0[0];
#pragma unroll
        for (int j = 0; j < 8; ++j) {
            mx = fmaxf(mx, s0[j]);
            mx = fmaxf(mx, s1[j]);
        }
        mx = fmaxf(mx, __shfl_xor(mx, 16, 32));
        float mnew  = fmaxf(mrow, mx);
        float alpha = __expf(mrow - mnew);
        float psum  = 0.0f;
        v16bf pa;                 // P already in A-operand layout
#pragma unroll
        for (int j = 0; j < 8; ++j) {
            float p0 = __expf(s0[j] - mnew);
            float p1 = __expf(s1[j] - mnew);
            psum += p0 + p1;
            pa[j]     = (__bf16)p0;   // keys 8*h2 + j
            pa[8 + j] = (__bf16)p1;   // keys 16 + 8*h2 + j
        }
        psum += __shfl_xor(psum, 16, 32);
        lrow = lrow * alpha + psum;
        mrow = mnew;

        // broadcast alpha per query row and rescale O (O lane = h column)
        float ar[8];
#pragma unroll
        for (int r = 0; r < 8; ++r) ar[r] = __shfl(alpha, r + 8 * h2, 32);
#pragma unroll
        for (int n = 0; n < 4; ++n)
#pragma unroll
            for (int r = 0; r < 8; ++r) O[n][r] *= ar[r];

        // ---- O += P @ V from LDS V^T tile ----
#pragma unroll
        for (int nt = 0; nt < 4; ++nt) {
            const __bf16* vp = &vsm[buf][(16 * nt + l15) * VROW + 8 * h2];
            v8bf vlo = *(const v8bf*)vp;
            v8bf vhi = *(const v8bf*)(vp + 16);
            v16bf vb;
#pragma unroll
            for (int j = 0; j < 8; ++j) { vb[j] = vlo[j]; vb[8 + j] = vhi[j]; }
            O[nt] = wmma_bf16(pa, vb, O[nt]);
        }
        __syncthreads();                      // reads done before overwrite
    }

    // normalize and store fp32 output [B,T,H]
    float rinv = 1.0f / lrow;
    float rr[8];
#pragma unroll
    for (int r = 0; r < 8; ++r) rr[r] = __shfl(rinv, r + 8 * h2, 32);
#pragma unroll
    for (int nt = 0; nt < 4; ++nt) {
        int hcol = 16 * nt + l15;
#pragma unroll
        for (int r = 0; r < 8; ++r)
            out[(g0 + r + 8 * h2) * H_ + hcol] = O[nt][r] * rr[r];
    }
}

// ---------------------------------------------------------------------------
extern "C" void kernel_launch(void* const* d_in, const int* in_sizes, int n_in,
                              void* d_out, int out_size, void* d_ws, size_t ws_size,
                              hipStream_t stream) {
    const float* x  = (const float*)d_in[0];
    const float* Wk = (const float*)d_in[1];
    const float* Wq = (const float*)d_in[2];
    const float* Wv = (const float*)d_in[3];
    float* out = (float*)d_out;

    // workspace layout (bytes)
    char* ws = (char*)d_ws;
    __bf16* wtq = (__bf16*)(ws + 0);                  // 128 KB
    __bf16* wtk = (__bf16*)(ws + 131072);             // 128 KB
    __bf16* wtv = (__bf16*)(ws + 262144);             // 128 KB
    __bf16* qb  = (__bf16*)(ws + 393216);             // 2 MB  [BT,H]
    __bf16* kbf = (__bf16*)(ws + 2490368);            // 2 MB  [BT,H]
    __bf16* vtb = (__bf16*)(ws + 4587520);            // 2 MB  [B,H,T]

    prep_weights<<<(3 * E_ * H_) / 256, 256, 0, stream>>>(Wk, Wq, Wv,
                                                          wtk, wtq, wtv);
    qkv_proj<<<BT_ / 128, 256, 0, stream>>>(x, wtq, wtk, wtv, qb, kbf, vtb);
    flash_attn<<<(BT_ / 16) / 8, 256, 0, stream>>>(qb, kbf, vtb, out);
}
